// GNNBiaffine_60936995995938
// MI455X (gfx1250) — matrix operand order repaired and checked by reading, not compiled, not run on hardware
//
#include <hip/hip_runtime.h>

typedef __attribute__((ext_vector_type(16))) __bf16 v16bf;
typedef __attribute__((ext_vector_type(8)))  float  v8f;
typedef __attribute__((ext_vector_type(4)))  int    v4i;

union ABfrag { v16bf v; uint4 q[2]; };
union Pack8  { uint4 q; __bf16 h[8]; };

#define TMr 128
#define TNr 64
#define TKr 32

// Async global->LDS staging (CDNA5 GLOBAL_LOAD_ASYNC_TO_LDS_B128, ASYNCcnt).
#if defined(__AMDGCN__) && __has_builtin(__builtin_amdgcn_global_load_async_to_lds_b128)
#define USE_ASYNC_LDS 1
#endif

// Pointer types matching the builtin signature: (v4i AS1*, v4i AS3*, imm, imm)
typedef __attribute__((address_space(1))) v4i* gv4p; // global int4*
typedef __attribute__((address_space(3))) v4i* lv4p; // LDS int4*

__device__ __forceinline__ void wait_async_zero() {
#if defined(__AMDGCN__)
#if __has_builtin(__builtin_amdgcn_s_wait_asynccnt)
  __builtin_amdgcn_s_wait_asynccnt(0);
#else
  asm volatile("s_wait_asynccnt 0" ::: "memory");
#endif
#endif
}

// ---------------------------------------------------------------------------
// Batched bf16 GEMM, C = A(MxK) * B(KxN), f32 accumulate, fused epilogues.
// B is ALWAYS N-major: element (k,n) at Bm[n*ldb + k]  -> contiguous LDS staging.
// Workgroup = 4 waves, 128x64 tile; each wave owns 32x64 (2 A frags x 4 B frags
// -> 8 wmma per K-step, each B fragment reused by 2 wmma).
// Epilogue MODE:
//   0: out_bf16[m*N+n]  = bf16(acc + xaddT[n*M+m])          (aggregate + self loop)
//   1: out_bf16[n*M+m]  = bf16(relu((acc+bias[n])*invd[m])) (transform, writes x^T)
//   2: out_bf16[m*N+n]  = bf16(acc)                         (r1 @ U)
//   3: out_f32 [m*N+n]  = acc                               (t @ r2^T -> score)
// ---------------------------------------------------------------------------
template <int MODE>
__global__ __launch_bounds__(128)
void gemm_bf16(const __bf16* __restrict__ A, long long strA, int lda,
               const __bf16* __restrict__ Bm, long long strB, int ldb,
               int M, int N, int K,
               const __bf16* __restrict__ xaddT, long long strX,
               const float* __restrict__ bias,
               const float* __restrict__ invd, long long strD,
               __bf16* __restrict__ outb, long long strOb,
               float* __restrict__ outf, long long strOf)
{
  __shared__ __attribute__((aligned(16))) __bf16 As[TMr * TKr]; // 8 KB, row-major
  __shared__ __attribute__((aligned(16))) __bf16 Bs[TNr * TKr]; // 4 KB, stored [n][k]

  const int tilesM = M / TMr;
  const int tm = blockIdx.x % tilesM;
  const int tn = blockIdx.x / tilesM;
  const int batch = blockIdx.y;
  const __bf16* Ab = A + (size_t)batch * strA;
  const __bf16* Bb = Bm + (size_t)batch * strB;

  const int tid  = threadIdx.x;
  const int lane = tid & 31;
  const int w    = tid >> 5;     // wave 0..3 -> rows w*32 .. w*32+31 of the tile
  const int half = lane >> 4;
  const int l16  = lane & 15;

  // Per-thread staging addresses (LDS side is a linear copy of the tile rows).
  const int an  = tid >> 1;             // B-stage row (n), 64 rows x 2 threads
  const int ach = (tid & 1) * 16;       // B-stage k-chunk

  v8f acc[8] = {};

  for (int k0 = 0; k0 < K; k0 += TKr) {
    const __bf16* gA = Ab + (size_t)(tm * TMr + tid) * lda + k0;
    const __bf16* gB = Bb + (size_t)(tn * TNr + an) * ldb + (k0 + ach);
    __bf16* dA = &As[tid * TKr];
    __bf16* dB = &Bs[an * TKr + ach];
    __builtin_prefetch(gA + TKr, 0, 1);
    __builtin_prefetch(gB + TKr, 0, 1);
#if defined(USE_ASYNC_LDS)
    // DMA the tile straight into LDS: no VGPR round-trip, tracked by ASYNCcnt.
    __builtin_amdgcn_global_load_async_to_lds_b128((gv4p)gA, (lv4p)dA,  0, 0);
    __builtin_amdgcn_global_load_async_to_lds_b128((gv4p)gA, (lv4p)dA, 16, 0);
    __builtin_amdgcn_global_load_async_to_lds_b128((gv4p)gA, (lv4p)dA, 32, 0);
    __builtin_amdgcn_global_load_async_to_lds_b128((gv4p)gA, (lv4p)dA, 48, 0);
    __builtin_amdgcn_global_load_async_to_lds_b128((gv4p)gB, (lv4p)dB,  0, 0);
    __builtin_amdgcn_global_load_async_to_lds_b128((gv4p)gB, (lv4p)dB, 16, 0);
    wait_async_zero();
#else
    {
      const uint4 u0 = ((const uint4*)gA)[0];
      const uint4 u1 = ((const uint4*)gA)[1];
      const uint4 u2 = ((const uint4*)gA)[2];
      const uint4 u3 = ((const uint4*)gA)[3];
      uint4* d = (uint4*)dA;
      d[0] = u0; d[1] = u1; d[2] = u2; d[3] = u3;
      const uint4 b0 = ((const uint4*)gB)[0];
      const uint4 b1 = ((const uint4*)gB)[1];
      uint4* e = (uint4*)dB;
      e[0] = b0; e[1] = b1;
    }
#endif
    __syncthreads();

    // ---- A fragments: lane holds row m, k = (ev/8)*16 + half*8 + ev%8 ----
    ABfrag a0, a1;
    {
      const __bf16* base0 = &As[(w * 32 + l16) * TKr + half * 8];
      a0.q[0] = *(const uint4*)(base0);
      a0.q[1] = *(const uint4*)(base0 + 16);
      const __bf16* base1 = &As[(w * 32 + 16 + l16) * TKr + half * 8];
      a1.q[0] = *(const uint4*)(base1);
      a1.q[1] = *(const uint4*)(base1 + 16);
    }
#pragma unroll
    for (int t = 0; t < 4; ++t) {
      // ---- B fragment: lane holds col n, k = (ev/8)*16 + half*8 + ev%8 ----
      ABfrag bf;
      const __bf16* base = &Bs[(t * 16 + l16) * TKr + half * 8];
      bf.q[0] = *(const uint4*)(base);
      bf.q[1] = *(const uint4*)(base + 16);
      acc[t] = __builtin_amdgcn_wmma_f32_16x16x32_bf16(
          false, a0.v, false, bf.v, (short)0, acc[t], false, false);
      acc[4 + t] = __builtin_amdgcn_wmma_f32_16x16x32_bf16(
          false, a1.v, false, bf.v, (short)0, acc[4 + t], false, false);
    }
    __syncthreads();
  }

  // ---- epilogue: C/D layout: VGPR r -> m = r + 8*half, col n = l16 ----
#pragma unroll
  for (int p = 0; p < 2; ++p) {
    const int gm0 = tm * TMr + w * 32 + p * 16 + 8 * half; // multiple of 8
#pragma unroll
    for (int t = 0; t < 4; ++t) {
      const int gn = tn * TNr + t * 16 + l16;
      if (MODE == 0) {
        // self-loop add: 8 consecutive m at fixed n in x^T -> one b128 load
        Pack8 xl;
        xl.q = *(const uint4*)(xaddT + (size_t)batch * strX + (size_t)gn * M + gm0);
#pragma unroll
        for (int r = 0; r < 8; ++r) {
          const float v = acc[p * 4 + t][r] + (float)xl.h[r];
          outb[(size_t)batch * strOb + (size_t)(gm0 + r) * N + gn] = (__bf16)v;
        }
      } else if (MODE == 1) {
        // transform: write x^T, 8 consecutive m pack into one b128 store
        Pack8 pk;
#pragma unroll
        for (int r = 0; r < 8; ++r) {
          float v = (acc[p * 4 + t][r] + bias[gn]) *
                    invd[(size_t)batch * strD + gm0 + r];
          v = v > 0.f ? v : 0.f;
          pk.h[r] = (__bf16)v;
        }
        *(uint4*)(outb + (size_t)batch * strOb + (size_t)gn * M + gm0) = pk.q;
      } else if (MODE == 2) {
#pragma unroll
        for (int r = 0; r < 8; ++r)
          outb[(size_t)batch * strOb + (size_t)(gm0 + r) * N + gn] =
              (__bf16)acc[p * 4 + t][r];
      } else {
#pragma unroll
        for (int r = 0; r < 8; ++r)
          outf[(size_t)batch * strOf + (size_t)(gm0 + r) * N + gn] =
              acc[p * 4 + t][r];
      }
    }
  }
}

// ---------------------------------------------------------------------------
// Prep kernels
// ---------------------------------------------------------------------------
__global__ void prune_adj_kernel(const float* __restrict__ adj,
                                 __bf16* __restrict__ adjF,
                                 __bf16* __restrict__ adjB, int S)
{
  const int i = blockIdx.x * blockDim.x + threadIdx.x;
  const int SS = S * S;
  const int b = i / SS;
  const int rem = i - b * SS;
  const int x = rem / S;
  const int y = rem - x * S;
  const float v = adj[i];
  const float keep = (v >= 0.1f && v <= 0.9f) ? v : 0.0f;
  adjF[i] = (__bf16)keep;
  adjB[(size_t)b * SS + (size_t)y * S + x] = (__bf16)keep;
}

__global__ void denom_row_kernel(const float* __restrict__ adj,
                                 float* __restrict__ invd, int S)
{
  const int row = blockIdx.x; // b*S + x
  const float* p = adj + (size_t)row * S;
  float s = 0.f;
  for (int y = threadIdx.x; y < S; y += 256) {
    const float v = p[y];
    if (v >= 0.1f && v <= 0.9f) s += v;
  }
  __shared__ float red[256];
  red[threadIdx.x] = s;
  __syncthreads();
  for (int off = 128; off; off >>= 1) {
    if (threadIdx.x < off) red[threadIdx.x] += red[threadIdx.x + off];
    __syncthreads();
  }
  if (threadIdx.x == 0) invd[row] = 1.0f / (red[0] + 1.0f);
}

__global__ void denom_col_kernel(const float* __restrict__ adj,
                                 float* __restrict__ invd, int S)
{
  const int b = blockIdx.y;
  const int y = blockIdx.x * 256 + threadIdx.x;
  const float* p = adj + (size_t)b * S * S;
  float s = 0.f;
  for (int x = 0; x < S; ++x) {
    const float v = p[(size_t)x * S + y];
    if (v >= 0.1f && v <= 0.9f) s += v;
  }
  invd[b * S + y] = 1.0f / (s + 1.0f);
}

// seq [B,S,D] f32 -> x1^T, x2^T [B,D,S] bf16
__global__ void cvt_xT_kernel(const float* __restrict__ s,
                              __bf16* __restrict__ d1, __bf16* __restrict__ d2,
                              int S, int D)
{
  const int i = blockIdx.x * blockDim.x + threadIdx.x;
  const int SD = S * D;
  const int b = i / SD;
  const int rem = i - b * SD;
  const int ss = rem / D;
  const int dd = rem - ss * D;
  const __bf16 v = (__bf16)s[i];
  const size_t o = (size_t)b * SD + (size_t)dd * S + ss;
  d1[o] = v;
  d2[o] = v;
}

// W [L,D,D] f32 -> W^T [L,D,D] bf16  (wt[l,e,d] = w[l,d,e])
__global__ void cvt_WT_kernel(const float* __restrict__ w,
                              __bf16* __restrict__ wt, int D)
{
  const int i = blockIdx.x * blockDim.x + threadIdx.x;
  const int DD = D * D;
  const int l = i / DD;
  const int rem = i - l * DD;
  const int d = rem / D;
  const int e = rem - d * D;
  wt[(size_t)l * DD + (size_t)e * D + d] = (__bf16)w[i];
}

// U [Din,Din] f32 -> U^T padded [DP,DP] bf16  (Up[j*DP+i] = U[i*Din+j])
__global__ void pack_UT_kernel(const float* __restrict__ U,
                               __bf16* __restrict__ Up, int Din, int DP)
{
  const int i = blockIdx.x * blockDim.x + threadIdx.x; // over DP*DP
  const int j = i / DP;     // output row  (n of GEMM)
  const int k = i - j * DP; // output col  (k of GEMM)
  const float v = (j < Din && k < Din) ? U[(size_t)k * Din + j] : 0.f;
  Up[i] = (__bf16)v;
}

// x^T [B,D,S] bf16 -> r [B,S,DP] row-major with [h,1,0...] padding
__global__ void pack_r_kernel(const __bf16* __restrict__ x1T,
                              const __bf16* __restrict__ x2T,
                              __bf16* __restrict__ r1,
                              __bf16* __restrict__ r2,
                              int S, int D, int DP)
{
  const int i = blockIdx.x * blockDim.x + threadIdx.x; // over B*S*DP
  const int row = i / DP;   // b*S + s
  const int d = i - row * DP;
  const int b = row / S;
  const int s = row - b * S;
  const __bf16 one = (__bf16)1.0f;
  const __bf16 zero = (__bf16)0.0f;
  const size_t src = (size_t)b * S * D + (size_t)d * S + s;
  r1[i] = (d < D) ? x1T[src] : (d == D ? one : zero);
  r2[i] = (d < D) ? x2T[src] : (d == D ? one : zero);
}

// ---------------------------------------------------------------------------
template <int MODE>
static void run_gemm(hipStream_t stream, int Bn,
                     const __bf16* A, long long strA, int lda,
                     const __bf16* Bm, long long strB, int ldb,
                     int M, int N, int K,
                     const __bf16* xaddT, long long strX,
                     const float* bias, const float* invd, long long strD,
                     __bf16* outb, long long strOb,
                     float* outf, long long strOf)
{
  dim3 grid((M / TMr) * (N / TNr), Bn);
  gemm_bf16<MODE><<<grid, dim3(128), 0, stream>>>(
      A, strA, lda, Bm, strB, ldb, M, N, K,
      xaddT, strX, bias, invd, strD, outb, strOb, outf, strOf);
}

extern "C" void kernel_launch(void* const* d_in, const int* in_sizes, int n_in,
                              void* d_out, int out_size, void* d_ws, size_t ws_size,
                              hipStream_t stream)
{
  (void)in_sizes; (void)n_in; (void)out_size; (void)ws_size;
  constexpr int Bn = 8, S = 1024, D = 512, L = 2, DP = 576; // DP = pad(D+1) to 64
  constexpr int Din1 = D + 1;

  const float* seq  = (const float*)d_in[0];
  const float* adj  = (const float*)d_in[1];
  const float* W1f  = (const float*)d_in[2];
  const float* b1f  = (const float*)d_in[3];
  const float* W2f  = (const float*)d_in[4];
  const float* b2f  = (const float*)d_in[5];
  const float* Uf   = (const float*)d_in[6];
  float* out = (float*)d_out;

  // workspace carve-up (all sizes multiples of 256 B)
  char* p = (char*)d_ws;
  auto carve = [&](size_t bytes) { void* r = (void*)p; p += (bytes + 255) & ~(size_t)255; return r; };
  __bf16* adjF  = (__bf16*)carve((size_t)Bn * S * S * 2);
  __bf16* adjB  = (__bf16*)carve((size_t)Bn * S * S * 2);
  __bf16* x1T   = (__bf16*)carve((size_t)Bn * S * D * 2);  // [B,D,S]
  __bf16* x2T   = (__bf16*)carve((size_t)Bn * S * D * 2);  // [B,D,S]
  __bf16* Axpx  = (__bf16*)carve((size_t)Bn * S * D * 2);  // [B,S,D]
  __bf16* W1t   = (__bf16*)carve((size_t)L * D * D * 2);   // [L,D,D] transposed
  __bf16* W2t   = (__bf16*)carve((size_t)L * D * D * 2);
  __bf16* Upt   = (__bf16*)carve((size_t)DP * DP * 2);     // U^T padded
  __bf16* r1p   = (__bf16*)carve((size_t)Bn * S * DP * 2); // [B,S,DP]
  __bf16* r2p   = (__bf16*)carve((size_t)Bn * S * DP * 2);
  __bf16* tbuf  = (__bf16*)carve((size_t)Bn * S * DP * 2);
  float*  invdF = (float*)carve((size_t)Bn * S * 4);
  float*  invdB = (float*)carve((size_t)Bn * S * 4);

  // --- prep ---
  cvt_xT_kernel<<<(Bn * S * D) / 256, 256, 0, stream>>>(seq, x1T, x2T, S, D);
  prune_adj_kernel<<<(Bn * S * S) / 256, 256, 0, stream>>>(adj, adjF, adjB, S);
  denom_row_kernel<<<Bn * S, 256, 0, stream>>>(adj, invdF, S);
  denom_col_kernel<<<dim3(S / 256, Bn), 256, 0, stream>>>(adj, invdB, S);
  cvt_WT_kernel<<<(L * D * D) / 256, 256, 0, stream>>>(W1f, W1t, D);
  cvt_WT_kernel<<<(L * D * D) / 256, 256, 0, stream>>>(W2f, W2t, D);
  pack_UT_kernel<<<(DP * DP) / 256, 256, 0, stream>>>(Uf, Upt, Din1, DP);

  // --- two GCN streams, L layers each ---
  for (int l = 0; l < L; ++l) {
    // stream 1 (forward adjacency)
    run_gemm<0>(stream, Bn, adjF, (long long)S * S, S,
                x1T, (long long)S * D, S, S, D, S,       // B = x^T, ldb = S
                x1T, (long long)S * D, nullptr, nullptr, 0,
                Axpx, (long long)S * D, nullptr, 0);
    run_gemm<1>(stream, Bn, Axpx, (long long)S * D, D,
                W1t + (size_t)l * D * D, 0, D, S, D, D,  // B = W^T, ldb = D
                nullptr, 0, b1f + l * D, invdF, (long long)S,
                x1T, (long long)S * D, nullptr, 0);       // writes x^T
    // stream 2 (backward adjacency = adj^T)
    run_gemm<0>(stream, Bn, adjB, (long long)S * S, S,
                x2T, (long long)S * D, S, S, D, S,
                x2T, (long long)S * D, nullptr, nullptr, 0,
                Axpx, (long long)S * D, nullptr, 0);
    run_gemm<1>(stream, Bn, Axpx, (long long)S * D, D,
                W2t + (size_t)l * D * D, 0, D, S, D, D,
                nullptr, 0, b2f + l * D, invdB, (long long)S,
                x2T, (long long)S * D, nullptr, 0);
  }

  // --- biaffine ---
  pack_r_kernel<<<(Bn * S * DP) / 256, 256, 0, stream>>>(x1T, x2T, r1p, r2p, S, D, DP);
  // t = r1 @ U            [S x DP] = [S x DP][DP x DP], B = U^T (N-major)
  run_gemm<2>(stream, Bn, r1p, (long long)S * DP, DP,
              Upt, 0, DP, S, DP, DP,
              nullptr, 0, nullptr, nullptr, 0,
              tbuf, (long long)S * DP, nullptr, 0);
  // score = t @ r2^T      [S x S]  = [S x DP][DP x S], B = r2 rows (N-major)
  run_gemm<3>(stream, Bn, tbuf, (long long)S * DP, DP,
              r2p, (long long)S * DP, DP, S, S, DP,
              nullptr, 0, nullptr, nullptr, 0,
              nullptr, 0, out, (long long)S * S);
}
